// NeuralRetriever_46935402611173
// MI455X (gfx1250) — matrix operand order repaired and checked by reading, not compiled, not run on hardware
//
#include <hip/hip_runtime.h>

// ---------------------------------------------------------------------------
// Problem constants (match reference setup_inputs)
// ---------------------------------------------------------------------------
#define NQ   64
#define QTOK 32
#define ND   50000
#define DTOK 16
#define EMB  768
#define HID  512
#define TOPK 8

// ---------------------------------------------------------------------------
// Types for CDNA5 WMMA (wave32): v_wmma_f32_16x16x32_bf16
// ---------------------------------------------------------------------------
typedef __bf16 bf16_t;
typedef bf16_t v16bf __attribute__((ext_vector_type(16)));
typedef float  v8f   __attribute__((ext_vector_type(8)));

struct alignas(16) I4 { int x, y, z, w; };     // POD 16-byte chunk (b128)
union Frag { v16bf v; I4 q[2]; };              // 32 bytes = 16 bf16

__device__ __forceinline__ unsigned short f32_to_bf16(float f) {
  unsigned int u = __float_as_uint(f);
  u += 0x7FFFu + ((u >> 16) & 1u);             // round-to-nearest-even
  return (unsigned short)(u >> 16);
}

// Async copy of one 16-byte chunk global -> LDS (ASYNCcnt-tracked, CDNA5).
__device__ __forceinline__ void async_b128_to_lds(const unsigned short* gsrc,
                                                  const unsigned short* lds_dst) {
  unsigned long long ga = (unsigned long long)(uintptr_t)gsrc;
  unsigned int       la = (unsigned int)(uintptr_t)lds_dst;  // low 32 bits = LDS addr
  asm volatile("global_load_async_to_lds_b128 %0, %1, off"
               :: "v"(la), "v"(ga) : "memory");
}
__device__ __forceinline__ void wait_asynccnt0() {
  asm volatile("s_wait_asynccnt 0x0" ::: "memory");
}

// ---------------------------------------------------------------------------
// Mean-pool tokens -> bf16  (one thread per [row, dim] output element)
// ---------------------------------------------------------------------------
__global__ void pool_to_bf16_k(const float* __restrict__ X, int rows, int toks,
                               int dim, unsigned short* __restrict__ Out) {
  int i = blockIdx.x * blockDim.x + threadIdx.x;
  if (i >= rows * dim) return;
  int r = i / dim, c = i - r * dim;
  const float* p = X + (size_t)r * toks * dim + c;
  float s = 0.f;
  for (int t = 0; t < toks; ++t) s += p[(size_t)t * dim];
  Out[i] = f32_to_bf16(s * (1.0f / (float)toks));
}

// ---------------------------------------------------------------------------
// W[K][N] fp32  ->  Wt[N][K] bf16  (B-fragment friendly layout)
// ---------------------------------------------------------------------------
__global__ void transpose_bf16_k(const float* __restrict__ W, int K, int N,
                                 unsigned short* __restrict__ Wt) {
  int i = blockIdx.x * blockDim.x + threadIdx.x;
  if (i >= K * N) return;
  int n = i / K, k = i - n * K;
  Wt[i] = f32_to_bf16(W[(size_t)k * N + n]);
}

// ---------------------------------------------------------------------------
// Fused GEMM (N fixed = 512) + epilogue.
//   EPI=0: ReLU(acc + bias)            -> bf16
//   EPI=1: LayerNorm(acc + bias; g,b)  -> bf16
//   EPI=2: L2-normalize(acc + bias)    -> bf16
// Block = 256 threads = 8 waves; block tile = 16 (M) x 512 (N);
// each wave owns 4 16x16 WMMA tiles.
// The full 16 x KK A-tile is async-staged into LDS ONCE, then the K-loop
// runs barrier-free: 2 ds_load_b128 + 8 global_load_b128 + 4 v_wmma per step.
// ---------------------------------------------------------------------------
template <int EPI, int KK>
__global__ void gemm512_wmma_k(const unsigned short* __restrict__ A, int M,
                               const unsigned short* __restrict__ Wt,   // [512][KK] bf16
                               const float* __restrict__ bias,          // [512]
                               const float* __restrict__ gamma,         // [512] (EPI==1)
                               const float* __restrict__ beta,          // [512] (EPI==1)
                               unsigned short* __restrict__ Out) {      // [M][512] bf16
  __shared__ alignas(16) unsigned short sA[16 * KK];   // 16/24 KB A tile
  __shared__ alignas(16) float sH[16 * 512];           // 32 KB epilogue spill
  __shared__ float sStat[16][2];

  const int wave = threadIdx.x >> 5;
  const int lane = threadIdx.x & 31;
  const int half = lane >> 4;       // K-half selector per ISA 16-bit layouts
  const int nloc = lane & 15;       // matrix row (A) / col (B,C) within tile
  const int m0   = blockIdx.x * 16;

  // ---- one-time async stage of the 16 x KK A-tile ----
  {
    constexpr int CPR = KK / 8;                 // b128 chunks per row
    for (int t = threadIdx.x; t < 16 * CPR; t += 256) {
      int r = t / CPR, c = t - r * CPR;
      int m = m0 + r;
      if (m >= M) m = M - 1;                    // clamp (M is a multiple of 16 here)
      async_b128_to_lds(A + (size_t)m * KK + c * 8, sA + t * 8);
    }
    wait_asynccnt0();
  }
  __syncthreads();

  const v8f vzero = {0.f, 0.f, 0.f, 0.f, 0.f, 0.f, 0.f, 0.f};
  v8f acc[4];
#pragma unroll
  for (int j = 0; j < 4; ++j) acc[j] = vzero;

#pragma unroll
  for (int ks = 0; ks < KK; ks += 32) {
    // A fragment: lane holds row m=nloc; VGPRs 0-3: K = ks+half*16..+7,
    //             VGPRs 4-7: K = ks+half*16+8..+15  (two b128 LDS reads)
    Frag a;
    a.q[0] = *(const I4*)(sA + nloc * KK + ks + half * 16);
    a.q[1] = *(const I4*)(sA + nloc * KK + ks + half * 16 + 8);

#pragma unroll
    for (int j = 0; j < 4; ++j) {
      int n = (wave * 4 + j) * 16 + nloc;
      // B fragment: lane holds col n; K = ks + half*16 .. +15 contiguous in Wt
      const I4* bp = (const I4*)(Wt + (size_t)n * KK + ks + half * 16);
      Frag b;
      b.q[0] = bp[0];
      b.q[1] = bp[1];
      acc[j] = __builtin_amdgcn_wmma_f32_16x16x32_bf16(
          false, a.v, false, b.v, (short)0, acc[j], false, false);
    }
  }

  if (EPI == 0) {
#pragma unroll
    for (int j = 0; j < 4; ++j) {
      int n = (wave * 4 + j) * 16 + nloc;
      float bn = bias[n];
#pragma unroll
      for (int v = 0; v < 8; ++v) {
        int m = m0 + half * 8 + v;                // C layout: m = half*8 + vgpr
        if (m < M) {
          float y = fmaxf(acc[j][v] + bn, 0.f);
          Out[(size_t)m * 512 + n] = f32_to_bf16(y);
        }
      }
    }
  } else {
#pragma unroll
    for (int j = 0; j < 4; ++j) {
      int n = (wave * 4 + j) * 16 + nloc;
      float bn = bias[n];
#pragma unroll
      for (int v = 0; v < 8; ++v)
        sH[(half * 8 + v) * 512 + n] = acc[j][v] + bn;
    }
    __syncthreads();
    {
      // per-row stats: wave w -> rows 2w (lanes 0-15) and 2w+1 (lanes 16-31)
      int r = 2 * wave + half;
      float sum = 0.f, sq = 0.f;
      for (int i = 0; i < 32; ++i) {
        float h = sH[r * 512 + nloc + 16 * i];
        sum += h;
        sq  += h * h;
      }
      for (int msk = 8; msk >= 1; msk >>= 1) {
        sum += __shfl_xor(sum, msk, 32);
        sq  += __shfl_xor(sq,  msk, 32);
      }
      if (nloc == 0) {
        if (EPI == 1) {
          float mu  = sum * (1.f / 512.f);
          float var = sq * (1.f / 512.f) - mu * mu;
          sStat[r][0] = mu;
          sStat[r][1] = rsqrtf(var + 1e-5f);
        } else {
          sStat[r][0] = 0.f;
          sStat[r][1] = 1.f / fmaxf(sqrtf(sq), 1e-12f);
        }
      }
    }
    __syncthreads();
    for (int i = 0; i < 32; ++i) {
      int p = threadIdx.x + 256 * i;      // 16*512 = 256*32 elements
      int r = p >> 9, c = p & 511;
      int m = m0 + r;
      if (m < M) {
        float h = sH[p];
        float y = (EPI == 1)
                      ? (h - sStat[r][0]) * sStat[r][1] * gamma[c] + beta[c]
                      : h * sStat[r][1];
        Out[(size_t)m * 512 + c] = f32_to_bf16(y);
      }
    }
  }
}

// ---------------------------------------------------------------------------
// Scores: S[m][n] = (Qn[m,:] . Dn[n,:]) / T.  Dn row-major == B layout.
// Block = 8 waves, each one 16x16 tile -> block tile 16 queries x 128 docs.
// The 16x512 Q-tile is async-staged once; K-loop is barrier-free.
// ---------------------------------------------------------------------------
__global__ void scores_wmma_k(const unsigned short* __restrict__ Qn,   // [64][512]
                              const unsigned short* __restrict__ Dn,   // [Nd][512]
                              int Nd, const float* __restrict__ tptr,
                              float* __restrict__ S) {                 // [64][Nd]
  __shared__ alignas(16) unsigned short sA[16 * 512];  // 16 KB Q tile
  const int wave = threadIdx.x >> 5;
  const int lane = threadIdx.x & 31;
  const int half = lane >> 4;
  const int nloc = lane & 15;
  const int m0 = blockIdx.x * 16;
  const int n  = blockIdx.y * 128 + wave * 16 + nloc;
  const int nC = (n < Nd) ? n : (Nd - 1);           // clamp OOB fragment loads
  constexpr int K = 512;

  {
    constexpr int CPR = K / 8;
    for (int t = threadIdx.x; t < 16 * CPR; t += 256) {
      int r = t / CPR, c = t - r * CPR;
      async_b128_to_lds(Qn + (size_t)(m0 + r) * K + c * 8, sA + t * 8);
    }
    wait_asynccnt0();
  }
  __syncthreads();

  v8f acc = {0.f, 0.f, 0.f, 0.f, 0.f, 0.f, 0.f, 0.f};

#pragma unroll
  for (int ks = 0; ks < K; ks += 32) {
    Frag a, b;
    a.q[0] = *(const I4*)(sA + nloc * K + ks + half * 16);
    a.q[1] = *(const I4*)(sA + nloc * K + ks + half * 16 + 8);
    const I4* bp = (const I4*)(Dn + (size_t)nC * K + ks + half * 16);
    b.q[0] = bp[0];
    b.q[1] = bp[1];
    acc = __builtin_amdgcn_wmma_f32_16x16x32_bf16(
        false, a.v, false, b.v, (short)0, acc, false, false);
  }

  float invT = 1.f / tptr[0];
#pragma unroll
  for (int v = 0; v < 8; ++v) {
    int m = m0 + half * 8 + v;
    if (n < Nd && m < NQ) S[(size_t)m * Nd + n] = acc[v] * invT;
  }
}

// ---------------------------------------------------------------------------
// Top-8 per query: per-thread sorted-8, LDS merge by thread 0.
// ---------------------------------------------------------------------------
__global__ void topk_k(const float* __restrict__ S, int Nd,
                       float* __restrict__ outScores, int* __restrict__ outIdx) {
  __shared__ float ss[256 * TOPK];
  __shared__ int   si[256 * TOPK];
  const int q = blockIdx.x;
  const int tid = threadIdx.x;

  float bs[TOPK];
  int   bi[TOPK];
#pragma unroll
  for (int j = 0; j < TOPK; ++j) { bs[j] = -3.4e38f; bi[j] = 0; }

  for (int d = tid; d < Nd; d += 256) {
    float v = S[(size_t)q * Nd + d];
    if (v > bs[TOPK - 1]) {
      int j = TOPK - 1;
      while (j > 0 && bs[j - 1] < v) { bs[j] = bs[j - 1]; bi[j] = bi[j - 1]; --j; }
      bs[j] = v; bi[j] = d;
    }
  }
#pragma unroll
  for (int j = 0; j < TOPK; ++j) { ss[tid * TOPK + j] = bs[j]; si[tid * TOPK + j] = bi[j]; }
  __syncthreads();

  if (tid == 0) {
    for (int j = 0; j < TOPK; ++j) {
      float best = -3.4e38f; int bidx = 0, bpos = 0;
      for (int t = 0; t < 256 * TOPK; ++t)
        if (ss[t] > best) { best = ss[t]; bidx = si[t]; bpos = t; }
      ss[bpos] = -3.4e38f;
      outScores[q * TOPK + j] = best;
      outIdx[q * TOPK + j]    = bidx;
    }
  }
}

// ---------------------------------------------------------------------------
// Gather winning doc embeddings: out[b] = docs[idx[b]]  (b = q*8+k), float4 copy.
// ---------------------------------------------------------------------------
__global__ void gather_k(const float* __restrict__ docs, const int* __restrict__ idx,
                         float* __restrict__ out) {
  const int b = blockIdx.x;
  const int doc = idx[b];
  const float4* src = (const float4*)(docs + (size_t)doc * DTOK * EMB);
  float4* dst = (float4*)(out + (size_t)b * DTOK * EMB);
  for (int i = threadIdx.x; i < (DTOK * EMB) / 4; i += blockDim.x) dst[i] = src[i];
}

// ---------------------------------------------------------------------------
// Launch
// ---------------------------------------------------------------------------
extern "C" void kernel_launch(void* const* d_in, const int* in_sizes, int n_in,
                              void* d_out, int out_size, void* d_ws, size_t ws_size,
                              hipStream_t stream) {
  (void)in_sizes; (void)n_in; (void)out_size; (void)ws_size;

  const float* qEmb = (const float*)d_in[0];
  const float* dEmb = (const float*)d_in[1];
  const float* qW1  = (const float*)d_in[2];
  const float* qb1  = (const float*)d_in[3];
  const float* qW2  = (const float*)d_in[4];
  const float* qb2  = (const float*)d_in[5];
  const float* qg   = (const float*)d_in[6];
  const float* qbe  = (const float*)d_in[7];
  const float* dW1  = (const float*)d_in[8];
  const float* db1  = (const float*)d_in[9];
  const float* dW2  = (const float*)d_in[10];
  const float* db2  = (const float*)d_in[11];
  const float* dg   = (const float*)d_in[12];
  const float* dbe  = (const float*)d_in[13];
  const float* pW   = (const float*)d_in[14];
  const float* pb   = (const float*)d_in[15];
  const float* temp = (const float*)d_in[16];
  float* out = (float*)d_out;

  char* ws = (char*)d_ws;
  size_t off = 0;
  auto take = [&](size_t bytes) {
    size_t r = off;
    off += (bytes + 255) & ~(size_t)255;
    return r;
  };
  size_t oDPool = take((size_t)ND * EMB * 2);  // bf16 pooled docs; later reused as fp32 scores
  size_t oQPool = take((size_t)NQ * EMB * 2);
  size_t oDH1   = take((size_t)ND * HID * 2);  // doc h1; later reused as d_norm
  size_t oDH2   = take((size_t)ND * HID * 2);
  size_t oQH1   = take((size_t)NQ * HID * 2);  // query h1; later reused as q_norm
  size_t oQH2   = take((size_t)NQ * HID * 2);
  size_t oQW1t  = take((size_t)HID * EMB * 2);
  size_t oDW1t  = take((size_t)HID * EMB * 2);
  size_t oQW2t  = take((size_t)HID * HID * 2);
  size_t oDW2t  = take((size_t)HID * HID * 2);
  size_t oPWt   = take((size_t)HID * HID * 2);
  size_t oIdx   = take((size_t)NQ * TOPK * 4);

  unsigned short* dPool = (unsigned short*)(ws + oDPool);
  unsigned short* qPool = (unsigned short*)(ws + oQPool);
  unsigned short* dH1   = (unsigned short*)(ws + oDH1);
  unsigned short* dH2   = (unsigned short*)(ws + oDH2);
  unsigned short* qH1   = (unsigned short*)(ws + oQH1);
  unsigned short* qH2   = (unsigned short*)(ws + oQH2);
  unsigned short* qW1t  = (unsigned short*)(ws + oQW1t);
  unsigned short* dW1t  = (unsigned short*)(ws + oDW1t);
  unsigned short* qW2t  = (unsigned short*)(ws + oQW2t);
  unsigned short* dW2t  = (unsigned short*)(ws + oDW2t);
  unsigned short* pWt   = (unsigned short*)(ws + oPWt);
  float* scoresBuf      = (float*)(ws + oDPool);   // overlays consumed dPool
  int*   topIdx         = (int*)(ws + oIdx);

  // Weight prep: fp32 [K][N] -> bf16 [N][K]
  transpose_bf16_k<<<(EMB * HID + 255) / 256, 256, 0, stream>>>(qW1, EMB, HID, qW1t);
  transpose_bf16_k<<<(EMB * HID + 255) / 256, 256, 0, stream>>>(dW1, EMB, HID, dW1t);
  transpose_bf16_k<<<(HID * HID + 255) / 256, 256, 0, stream>>>(qW2, HID, HID, qW2t);
  transpose_bf16_k<<<(HID * HID + 255) / 256, 256, 0, stream>>>(dW2, HID, HID, dW2t);
  transpose_bf16_k<<<(HID * HID + 255) / 256, 256, 0, stream>>>(pW,  HID, HID, pWt);

  // Token pooling -> bf16
  pool_to_bf16_k<<<(NQ * EMB + 255) / 256, 256, 0, stream>>>(qEmb, NQ, QTOK, EMB, qPool);
  pool_to_bf16_k<<<(ND * EMB + 255) / 256, 256, 0, stream>>>(dEmb, ND, DTOK, EMB, dPool);

  // Layer 1: relu(x @ W1 + b1)
  gemm512_wmma_k<0, EMB><<<(ND + 15) / 16, 256, 0, stream>>>(dPool, ND, dW1t, db1, nullptr, nullptr, dH1);
  gemm512_wmma_k<0, EMB><<<(NQ + 15) / 16, 256, 0, stream>>>(qPool, NQ, qW1t, qb1, nullptr, nullptr, qH1);
  // Layer 2: LN(h1 @ W2 + b2)
  gemm512_wmma_k<1, HID><<<(ND + 15) / 16, 256, 0, stream>>>(dH1, ND, dW2t, db2, dg, dbe, dH2);
  gemm512_wmma_k<1, HID><<<(NQ + 15) / 16, 256, 0, stream>>>(qH1, NQ, qW2t, qb2, qg, qbe, qH2);
  // Projection + L2 normalize (outputs overlay the now-free H1 buffers)
  gemm512_wmma_k<2, HID><<<(ND + 15) / 16, 256, 0, stream>>>(dH2, ND, pWt, pb, nullptr, nullptr, dH1);
  gemm512_wmma_k<2, HID><<<(NQ + 15) / 16, 256, 0, stream>>>(qH2, NQ, pWt, pb, nullptr, nullptr, qH1);

  // Similarity scores, top-k, gather
  scores_wmma_k<<<dim3(NQ / 16, (ND + 127) / 128), 256, 0, stream>>>(qH1, dH1, ND, temp, scoresBuf);
  topk_k<<<NQ, 256, 0, stream>>>(scoresBuf, ND, out, topIdx);
  gather_k<<<NQ * TOPK, 256, 0, stream>>>(dEmb, topIdx, out + NQ * TOPK);
}
